// ANI_26431228739595
// MI455X (gfx1250) — compile-verified, matrix operand order checked by reading, not run on hardware
//
#include <hip/hip_runtime.h>

// CDNA5 WMMA types
typedef __attribute__((ext_vector_type(16))) _Float16 v16h;
typedef __attribute__((ext_vector_type(8)))  float    v8f;

#define WAVES_PER_BLOCK 4
#define PAIRS_PER_TILE  1536   // 16 atoms * 96 neighbors, contiguous in r_ij/mask
#define N_NEI           96
#define R_RAD           16

// gfx1250 async global->LDS path (ASYNCcnt-tracked), with safe fallbacks.
// Probe round 2 showed the builtin's params are v4i pointers:
//   arg0: 'int __attribute__((vector_size(16))) __device__ *'  (addrspace(1))
//   arg1: LDS destination (addrspace(3)), args 2/3: imm offset, imm cpol.
#if defined(__AMDGCN__) && __has_builtin(__builtin_amdgcn_global_load_async_to_lds_b128)
#define HAVE_ASYNC_LDS 1
typedef int v4i __attribute__((vector_size(16)));
typedef __attribute__((address_space(1))) v4i as1_v4i;
typedef __attribute__((address_space(3))) v4i as3_v4i;
#endif

__device__ __forceinline__ void wait_async_all() {
#if defined(HAVE_ASYNC_LDS)
#if __has_builtin(__builtin_amdgcn_s_wait_asynccnt)
    __builtin_amdgcn_s_wait_asynccnt(0);
#else
    asm volatile("s_wait_asynccnt 0" ::: "memory");
#endif
#endif
}

// out[a, r] = sum_n exp(-eta_r*(d[a,n]-rs_r)^2) * cutoff(d) * mask
// Reduction over n via V_WMMA_F32_16X16X32_F16:
//   D[r, atom] += A[r, k] * B[k, atom];  A = f16 f-values, B = 0/1 atom-selection.
// 96 = 3*32 -> each K=32 chunk belongs to one atom, so B is a ones-column and
// the result is invariant to the pair->K-slot permutation inside a chunk.
__global__ __launch_bounds__(128)
void ani_rbf_wmma(const float* __restrict__ r_ij,
                  const float* __restrict__ mask,
                  const float* __restrict__ etas,
                  const float* __restrict__ rss,
                  float* __restrict__ out)
{
    __shared__ __align__(16) float s_d[WAVES_PER_BLOCK][PAIRS_PER_TILE];
    __shared__ __align__(16) float s_c[WAVES_PER_BLOCK][PAIRS_PER_TILE]; // mask -> cm

    const int lane = threadIdx.x & 31;          // wave32
    const int wave = threadIdx.x >> 5;
    const int waveGlobal = blockIdx.x * WAVES_PER_BLOCK + wave;
    const size_t tileBase = (size_t)waveGlobal * PAIRS_PER_TILE;
    const int atom0 = waveGlobal * 16;          // first atom of this tile

    // ---- Stage 1: global -> LDS, pure copy (async DMA path on CDNA5) ----
#if defined(HAVE_ASYNC_LDS)
    #pragma unroll
    for (int it = 0; it < 12; ++it) {
        const int idx = it * 128 + lane * 4;    // 16B per lane, 512B per wave/instr
        __builtin_amdgcn_global_load_async_to_lds_b128(
            (as1_v4i*)(r_ij + tileBase + idx), (as3_v4i*)&s_d[wave][idx], 0, 0);
        __builtin_amdgcn_global_load_async_to_lds_b128(
            (as1_v4i*)(mask + tileBase + idx), (as3_v4i*)&s_c[wave][idx], 0, 0);
    }
    asm volatile("" ::: "memory");
    wait_async_all();
    asm volatile("" ::: "memory");
#else
    #pragma unroll
    for (int it = 0; it < 12; ++it) {
        const int idx = it * 128 + lane * 4;
        *(float4*)&s_d[wave][idx] = *(const float4*)(r_ij + tileBase + idx);
        *(float4*)&s_c[wave][idx] = *(const float4*)(mask + tileBase + idx);
    }
#endif

    // ---- Stage 2: cm = mask * cosine_cutoff(d), in place over the mask ----
    // Computed ONCE per pair (16x fewer v_cos than the naive formulation).
    const float PI_OVER_RC = 3.14159265358979323846f / 3.0f;
    #pragma unroll
    for (int it = 0; it < 12; ++it) {
        const int idx = it * 128 + lane * 4;
        const float4 dv = *(const float4*)&s_d[wave][idx];
        const float4 mv = *(const float4*)&s_c[wave][idx];
        float4 cm;
        cm.x = (dv.x < 3.0f) ? mv.x * (0.5f * __cosf(dv.x * PI_OVER_RC) + 0.5f) : 0.0f;
        cm.y = (dv.y < 3.0f) ? mv.y * (0.5f * __cosf(dv.y * PI_OVER_RC) + 0.5f) : 0.0f;
        cm.z = (dv.z < 3.0f) ? mv.z * (0.5f * __cosf(dv.z * PI_OVER_RC) + 0.5f) : 0.0f;
        cm.w = (dv.w < 3.0f) ? mv.w * (0.5f * __cosf(dv.w * PI_OVER_RC) + 0.5f) : 0.0f;
        *(float4*)&s_c[wave][idx] = cm;
    }
    // DS ops complete in order within a wave (ISA: "LDS: in order"); the tile
    // is wave-private, so no workgroup barrier is needed — just stop the
    // compiler from reordering LDS accesses across this point.
    asm volatile("" ::: "memory");

    // ---- Stage 3: 48 WMMAs accumulate the 16x16 tile ----
    // Lane feeds A-matrix row M = lane&15 => fixed radial index r per lane.
    // exp(-eta*t^2) = exp2(-(a1*d + a0)^2): fma + neg-mul + v_exp_f32 per elem.
    const int rIdx = lane & 15;
    const float eta = etas[rIdx];
    const float rs  = rss[rIdx];
    const float a1  = sqrtf(eta * 1.4426950408889634f);   // sqrt(eta*log2(e))
    const float a0  = -rs * a1;
    const int half  = lane >> 4;     // lane half covers pairs {0..15} / {16..31}
    const int kOff  = half * 16;

#if __has_builtin(__builtin_amdgcn_exp2f)
    #define EXP2F(x) __builtin_amdgcn_exp2f(x)
#else
    #define EXP2F(x) __expf((x) * 0.6931471805599453f)
#endif

    v8f acc = {0.f, 0.f, 0.f, 0.f, 0.f, 0.f, 0.f, 0.f};

    for (int atom = 0; atom < 16; ++atom) {
        // Selection B: ones in column 'atom', zeros elsewhere.
        const _Float16 sel = (rIdx == atom) ? (_Float16)1.0f : (_Float16)0.0f;
        v16h bsel;
        #pragma unroll
        for (int j = 0; j < 16; ++j) bsel[j] = sel;

        #pragma unroll
        for (int cc = 0; cc < 3; ++cc) {                 // 3 chunks of K=32
            const int base = atom * N_NEI + cc * 32 + kOff;
            v16h av;
            #pragma unroll
            for (int q = 0; q < 4; ++q) {
                const float4 d4  = *(const float4*)&s_d[wave][base + q * 4];
                const float4 cm4 = *(const float4*)&s_c[wave][base + q * 4];
                float u;
                u = fmaf(d4.x, a1, a0); av[q*4+0] = (_Float16)(EXP2F(-u * u) * cm4.x);
                u = fmaf(d4.y, a1, a0); av[q*4+1] = (_Float16)(EXP2F(-u * u) * cm4.y);
                u = fmaf(d4.z, a1, a0); av[q*4+2] = (_Float16)(EXP2F(-u * u) * cm4.z);
                u = fmaf(d4.w, a1, a0); av[q*4+3] = (_Float16)(EXP2F(-u * u) * cm4.w);
            }
            // D[r, atom] += A x B   (f32 accumulate)
            acc = __builtin_amdgcn_wmma_f32_16x16x32_f16(
                      /*neg_a=*/false, av, /*neg_b=*/false, bsel,
                      /*c_mod=*/(short)0, acc, /*reuse_a=*/false, /*reuse_b=*/false);
        }
    }

    // ---- Store: C/D layout -> lane holds column n = lane&15 (= atom),
    // VGPR j = row m = half*8 + j (= radial index). out rows are contiguous
    // 16 floats per atom -> two b128 stores per lane, fully coalesced.
    float* op = out + ((size_t)(atom0 + (lane & 15))) * R_RAD + half * 8;
    float4 o0 = {acc[0], acc[1], acc[2], acc[3]};
    float4 o1 = {acc[4], acc[5], acc[6], acc[7]};
    *(float4*)(op)     = o0;
    *(float4*)(op + 4) = o1;
}

extern "C" void kernel_launch(void* const* d_in, const int* in_sizes, int n_in,
                              void* d_out, int out_size, void* d_ws, size_t ws_size,
                              hipStream_t stream) {
    const float* r_ij = (const float*)d_in[0];   // [B, A, N] f32
    const float* mask = (const float*)d_in[1];   // [B, A, N] f32
    const float* etas = (const float*)d_in[2];   // [R] f32
    const float* rss  = (const float*)d_in[3];   // [R] f32
    float* out = (float*)d_out;                  // [B, A, R] f32

    const int totalPairs = in_sizes[0];                       // B*A*N
    const int nWaves  = totalPairs / PAIRS_PER_TILE;          // one tile per wave
    const int nBlocks = nWaves / WAVES_PER_BLOCK;

    hipLaunchKernelGGL(ani_rbf_wmma, dim3(nBlocks), dim3(128), 0, stream,
                       r_ij, mask, etas, rss, out);
}
